// GCN_MC_23106924052860
// MI455X (gfx1250) — matrix operand order repaired and checked by reading, not compiled, not run on hardware
//
#include <hip/hip_runtime.h>

#define N_NODES 50000
#define N_EDGES 800000
#define DIM 128
#define LDA 136   // padded LDS row stride in halves (136*2B=272B -> +4 dwords/row, conflict-free)

typedef __attribute__((ext_vector_type(16))) _Float16 v16h;
typedef __attribute__((ext_vector_type(8)))  _Float16 v8h;
typedef __attribute__((ext_vector_type(8)))  float    v8f;

// ---------------------------------------------------------------------------
// Hardware f32 atomic add, device scope (avoids CAS-loop expansion).
// ---------------------------------------------------------------------------
__device__ __forceinline__ void atomic_add_f32_dev(float* p, float v) {
  asm volatile("global_atomic_add_f32 %0, %1, off scope:SCOPE_DEV"
               :: "v"(p), "v"(v)
               : "memory");
}

// ---------------------------------------------------------------------------
// Kernel 1: zero the aggregation buffer (50000*128 floats in d_ws).
// ---------------------------------------------------------------------------
__global__ void __launch_bounds__(256) zero_agg(float4* __restrict__ ws, int n4) {
  int i = blockIdx.x * blockDim.x + threadIdx.x;
  if (i < n4) ws[i] = float4{0.f, 0.f, 0.f, 0.f};
}

// ---------------------------------------------------------------------------
// Kernel 2: split W into f16 hi/lo once (Markidis split), reused by all blocks.
// ---------------------------------------------------------------------------
__global__ void __launch_bounds__(256)
split_W(const float* __restrict__ W, _Float16* __restrict__ whi,
        _Float16* __restrict__ wlo) {
  const int i = blockIdx.x * blockDim.x + threadIdx.x;   // 0 .. 16383
  const float f = W[i];
  const _Float16 h = (_Float16)f;
  whi[i] = h;
  wlo[i] = (_Float16)(f - (float)h);
}

// ---------------------------------------------------------------------------
// Kernel 3: edge scatter-add.  One wave32 per edge; each lane moves 4 floats.
// ---------------------------------------------------------------------------
__global__ void __launch_bounds__(256)
scatter_add(const float* __restrict__ x,
            const int*   __restrict__ src,
            const int*   __restrict__ dst,
            float*       __restrict__ agg) {
  const int edge = (blockIdx.x * blockDim.x + threadIdx.x) >> 5;
  const int lane = threadIdx.x & 31;
  if (edge >= N_EDGES) return;
  const int s = src[edge];
  const int d = dst[edge];
  const float4 v = *(const float4*)(x + (size_t)s * DIM + lane * 4);
  float* p = agg + (size_t)d * DIM + lane * 4;
  atomic_add_f32_dev(p + 0, v.x);
  atomic_add_f32_dev(p + 1, v.y);
  atomic_add_f32_dev(p + 2, v.z);
  atomic_add_f32_dev(p + 3, v.w);
}

// ---------------------------------------------------------------------------
// Build a v16h fragment from two 16-byte runs.
// ---------------------------------------------------------------------------
__device__ __forceinline__ v16h frag16(const _Float16* p0, const _Float16* p1) {
  const v8h a = *(const v8h*)p0;
  const v8h b = *(const v8h*)p1;
  return __builtin_shufflevector(a, b, 0, 1, 2, 3, 4, 5, 6, 7,
                                 8, 9, 10, 11, 12, 13, 14, 15);
}

// ---------------------------------------------------------------------------
// Kernel 4: out = relu(agg @ W^T) + x  via split-precision f16 WMMA.
// Block = one 16-row strip of M; 256 threads split the A strip to LDS once;
// each of the 8 waves owns one 16x16 column tile (N = 128 = 8*16), K = 4*32.
//
// A fragment (16x32 f16): lane half = lane>>4, row m = lane&15;
//   element j holds K = kt*32 + half*8 + (j<8 ? j : j+8)  -> 2 x ds_load 16B.
// B fragment (32x16 f16): lane n = lane&15;
//   element j holds K = kt*32 + half*8 + (j<8 ? j : j+8)  -> 2 x global 16B.
// C/D (16x16 f32, v8f): element r -> (M = r + half*8, N = lane&15).
// ---------------------------------------------------------------------------
__global__ void __launch_bounds__(256)
gemm_relu_residual(const float*    __restrict__ agg,
                   const _Float16* __restrict__ whi,
                   const _Float16* __restrict__ wlo,
                   const float*    __restrict__ x,
                   float*          __restrict__ out) {
  __shared__ __attribute__((aligned(16))) _Float16 sAhi[16 * LDA];
  __shared__ __attribute__((aligned(16))) _Float16 sAlo[16 * LDA];

  const int tid   = threadIdx.x;
  const int mTile = blockIdx.x;          // 0..3124

  // ---- cooperative split of the 16x128 A strip into LDS (once per block) ---
  {
    const int row = tid >> 4;            // 0..15
    const int col = (tid & 15) * 8;      // 0..120
    const float* ap = agg + ((size_t)mTile * 16 + row) * DIM + col;
    const float4 f0 = *(const float4*)(ap);
    const float4 f1 = *(const float4*)(ap + 4);
    const float f[8] = {f0.x, f0.y, f0.z, f0.w, f1.x, f1.y, f1.z, f1.w};
    v8h hi, lo;
#pragma unroll
    for (int j = 0; j < 8; ++j) {
      const _Float16 h = (_Float16)f[j];
      hi[j] = h;
      lo[j] = (_Float16)(f[j] - (float)h);
    }
    *(v8h*)(&sAhi[row * LDA + col]) = hi;
    *(v8h*)(&sAlo[row * LDA + col]) = lo;
  }
  __syncthreads();

  const int lane  = tid & 31;
  const int wv    = tid >> 5;            // column tile 0..7
  const int half_ = lane >> 4;
  const int r16   = lane & 15;

  const _Float16* Bhi = whi + (size_t)(wv * 16 + r16) * DIM;  // W row n = B col n
  const _Float16* Blo = wlo + (size_t)(wv * 16 + r16) * DIM;

  v8f c = {};

#pragma unroll
  for (int kt = 0; kt < 4; ++kt) {
    const int ka = kt * 32 + half_ * 8;
    const int kb = kt * 32 + half_ * 8;

    const v16h ah = frag16(&sAhi[r16 * LDA + ka], &sAhi[r16 * LDA + ka + 16]);
    const v16h al = frag16(&sAlo[r16 * LDA + ka], &sAlo[r16 * LDA + ka + 16]);
    const v16h bh = frag16(Bhi + kb, Bhi + kb + 16);
    const v16h bl = frag16(Blo + kb, Blo + kb + 16);

    // (ah+al)*(bh+bl) ~= ah*bh + al*bh + ah*bl  (lo*lo below fp32 ulp)
    c = __builtin_amdgcn_wmma_f32_16x16x32_f16(false, ah, false, bh,
                                               (short)0, c, false, false);
    c = __builtin_amdgcn_wmma_f32_16x16x32_f16(false, al, false, bh,
                                               (short)0, c, false, false);
    c = __builtin_amdgcn_wmma_f32_16x16x32_f16(false, ah, false, bl,
                                               (short)0, c, false, false);
  }

  // ---- epilogue: relu + residual, per C/D VGPR layout ----------------------
  const int n = wv * 16 + r16;
#pragma unroll
  for (int r = 0; r < 8; ++r) {
    const int m = mTile * 16 + r + half_ * 8;
    float v = c[r];
    v = v > 0.f ? v : 0.f;
    out[(size_t)m * DIM + n] = v + x[(size_t)m * DIM + n];
  }
}

// ---------------------------------------------------------------------------
extern "C" void kernel_launch(void* const* d_in, const int* in_sizes, int n_in,
                              void* d_out, int out_size, void* d_ws, size_t ws_size,
                              hipStream_t stream) {
  (void)in_sizes; (void)n_in; (void)out_size; (void)ws_size;

  const float* x   = (const float*)d_in[0];
  const float* W   = (const float*)d_in[1];
  const int*   src = (const int*)d_in[2];
  const int*   dst = (const int*)d_in[3];
  float*       out = (float*)d_out;

  // workspace layout: agg (25.6 MB f32) | Whi (32 KB f16) | Wlo (32 KB f16)
  float*    agg = (float*)d_ws;
  _Float16* whi = (_Float16*)((char*)d_ws + (size_t)N_NODES * DIM * sizeof(float));
  _Float16* wlo = whi + DIM * DIM;

  // 1) agg = 0
  zero_agg<<<(N_NODES * DIM / 4 + 255) / 256, 256, 0, stream>>>(
      (float4*)agg, N_NODES * DIM / 4);

  // 2) W -> (Whi, Wlo) f16 split (once)
  split_W<<<DIM * DIM / 256, 256, 0, stream>>>(W, whi, wlo);

  // 3) agg[dst] += x[src]   (one wave per edge -> 8 edges per 256-thread block)
  scatter_add<<<(N_EDGES * 32) / 256, 256, 0, stream>>>(x, src, dst, agg);

  // 4) out = relu(agg @ W^T) + x   (3125 blocks x 8 waves = 25000 16x16 tiles)
  gemm_relu_residual<<<N_NODES / 16, 256, 0, stream>>>(agg, whi, wlo, x, out);
}
